// TransformerBlock_12730283065843
// MI455X (gfx1250) — matrix operand order repaired and checked by reading, not compiled, not run on hardware
//
#include <hip/hip_runtime.h>
#include <cstdint>
#include <cstddef>

// ---------------------------------------------------------------------------
// CDNA5 / gfx1250 transformer block.
// bf16 WMMA GEMMs with async-to-LDS double-buffered staging + fused attention.
// ---------------------------------------------------------------------------

typedef __attribute__((ext_vector_type(16))) __bf16 v16bf;
typedef __attribute__((ext_vector_type(8)))  float  v8f;
typedef __attribute__((ext_vector_type(4)))  int    i32x4;

#define DIM_     1024
#define SEQ_     2048
#define NHEADS_  16
#define HD_      64

#if __has_builtin(__builtin_amdgcn_global_load_async_to_lds_b128)
#define ASYNC_LDS_COPY 1
#endif

// 16-byte global->LDS copy. Async (ASYNCcnt) when the toolchain exposes the
// gfx1250 builtin, else plain load + ds_store fallback.
// Builtin signature (probed): (global i32x4*, shared i32x4*, imm off, imm cpol)
__device__ __forceinline__ void cp16(const void* g, void* l) {
#ifdef ASYNC_LDS_COPY
  __builtin_amdgcn_global_load_async_to_lds_b128(
      (__attribute__((address_space(1))) i32x4*)(uintptr_t)g,
      (__attribute__((address_space(3))) i32x4*)(uintptr_t)(uint32_t)(uintptr_t)l,
      0, 0);
#else
  *(uint4*)l = *(const uint4*)g;
#endif
}

// Wait for this wave's async copies, then workgroup barrier.
__device__ __forceinline__ void stage_join() {
#ifdef ASYNC_LDS_COPY
#if __has_builtin(__builtin_amdgcn_s_wait_asynccnt)
  __builtin_amdgcn_s_wait_asynccnt(0);
#else
  asm volatile("s_wait_asynccnt 0" ::: "memory");
#endif
#endif
  __syncthreads();
}

__device__ __forceinline__ v8f wmma_bf16(v16bf a, v16bf b, v8f c) {
  return __builtin_amdgcn_wmma_f32_16x16x32_bf16(
      false, a, false, b, (short)0, c, false, false);
}

// A operand fragment from LDS tile stored [M][K] (stride bf16 elements, even).
// Contiguous k order, not the ISA table order (matches frag_b): lane<16 gets K 0..15, lane>=16 K 16..31.
__device__ __forceinline__ v16bf frag_a(const __bf16* t, int stride, int m0, int kbase) {
  int lane = threadIdx.x & 31;
  int grp  = lane >> 4;
  int m    = m0 + (lane & 15);
  const uint32_t* p = (const uint32_t*)(t + (size_t)m * stride + kbase);
  v16bf a;
#pragma unroll
  for (int q = 0; q < 8; ++q) {
    int di = q + 8 * grp;
    union { uint32_t u; __bf16 h[2]; } c;
    c.u = p[di];
    a[2 * q]     = c.h[0];
    a[2 * q + 1] = c.h[1];
  }
  return a;
}

// B operand fragment from LDS tile stored transposed [N][K] (stride mult of 8).
// Lanes 0-15: K=kbase..+15, lanes 16-31: K=kbase+16..+31 -> two ds_load_b128.
__device__ __forceinline__ v16bf frag_b(const __bf16* t, int stride, int n0, int kbase) {
  int lane = threadIdx.x & 31;
  int grp  = lane >> 4;
  int n    = n0 + (lane & 15);
  const uint4* p = (const uint4*)(t + (size_t)n * stride + kbase + 16 * grp);
  uint4 w0 = p[0];
  uint4 w1 = p[1];
  union { uint4 v; __bf16 h[8]; } c0, c1;
  c0.v = w0; c1.v = w1;
  v16bf b;
#pragma unroll
  for (int i = 0; i < 8; ++i) { b[i] = c0.h[i]; b[8 + i] = c1.h[i]; }
  return b;
}

// ---------------------------------------------------------------------------
// f32 -> bf16 elementwise convert (n multiple of 1024).
// ---------------------------------------------------------------------------
__global__ __launch_bounds__(256) void cvt_bf16_kernel(
    const float* __restrict__ in, __bf16* __restrict__ out, int n) {
  int i = (blockIdx.x * 256 + threadIdx.x) * 4;
  float4 f = *(const float4*)(in + i);
  out[i + 0] = (__bf16)f.x;
  out[i + 1] = (__bf16)f.y;
  out[i + 2] = (__bf16)f.z;
  out[i + 3] = (__bf16)f.w;
}

// ---------------------------------------------------------------------------
// f32 [K][N] -> bf16 [N][K] tiled transpose-convert (K,N multiples of 32).
// ---------------------------------------------------------------------------
__global__ __launch_bounds__(256) void transpose_cvt_kernel(
    const float* __restrict__ in, __bf16* __restrict__ out, int K, int N) {
  __shared__ float t[32][33];
  int k0 = blockIdx.y * 32;
  int n0 = blockIdx.x * 32;
  int r  = threadIdx.x >> 3;        // 0..31
  int c4 = (threadIdx.x & 7) * 4;   // 0..28
  float4 f = *(const float4*)(in + (size_t)(k0 + r) * N + n0 + c4);
  t[r][c4 + 0] = f.x; t[r][c4 + 1] = f.y;
  t[r][c4 + 2] = f.z; t[r][c4 + 3] = f.w;
  __syncthreads();
  __bf16* o = out + (size_t)(n0 + r) * K + k0 + c4;
  o[0] = (__bf16)t[c4 + 0][r];
  o[1] = (__bf16)t[c4 + 1][r];
  o[2] = (__bf16)t[c4 + 2][r];
  o[3] = (__bf16)t[c4 + 3][r];
}

// ---------------------------------------------------------------------------
// GEMM: C = epilogue(A[M][K](bf16) @ Bt[N][K]^T(bf16) + bias) * scale
// flags: 1=bias, 2=exact GELU, 4=bf16 out [M][N], 8=bf16 out in per-head
//        transposed V layout [(b,h,d)][s].
// Block tile 64x256, 8 waves (2x4), wave tile 32x64, K-step 32,
// double-buffered async LDS staging.
// ---------------------------------------------------------------------------
__global__ __launch_bounds__(256) void gemm_bf16_kernel(
    const __bf16* __restrict__ A, const __bf16* __restrict__ Bt,
    const float* __restrict__ bias, void* __restrict__ C,
    int M, int N, int K, int flags, float scale) {
  constexpr int AST = 40;   // 32 + 8 pad
  constexpr int BST = 40;
  __shared__ __bf16 sA[2][64 * AST];
  __shared__ __bf16 sB[2][256 * BST];

  int tid  = threadIdx.x;
  int lane = tid & 31;
  int wid  = tid >> 5;
  int wm   = wid >> 2;   // 0..1
  int wn   = wid & 3;    // 0..3
  int m0   = blockIdx.y * 64;
  int n0   = blockIdx.x * 256;

  int ar = tid >> 2;          // 0..63
  int ac = (tid & 3) * 8;     // 0,8,16,24

  v8f acc[2][4] = {};
  int nk = K / 32;

  // stage tile at k0 into buffer buf
  auto stage = [&](int buf, int k0) {
    cp16(A + (size_t)(m0 + ar) * K + k0 + ac, &sA[buf][ar * AST + ac]);
#pragma unroll
    for (int i = 0; i < 4; ++i) {
      int n = ar + i * 64;
      cp16(Bt + (size_t)(n0 + n) * K + k0 + ac, &sB[buf][n * BST + ac]);
    }
  };

  stage(0, 0);
  stage_join();

  for (int kt = 0; kt < nk; ++kt) {
    int buf = kt & 1;
    if (kt + 1 < nk) stage(buf ^ 1, (kt + 1) * 32);
    if (kt + 2 < nk) {
      __builtin_prefetch(A  + (size_t)(m0 + ar) * K + (kt + 2) * 32, 0, 0);
      __builtin_prefetch(Bt + (size_t)(n0 + ar) * K + (kt + 2) * 32, 0, 0);
    }
    v16bf a0 = frag_a(sA[buf], AST, wm * 32,      0);
    v16bf a1 = frag_a(sA[buf], AST, wm * 32 + 16, 0);
#pragma unroll
    for (int j = 0; j < 4; ++j) {
      v16bf b = frag_b(sB[buf], BST, wn * 64 + j * 16, 0);
      acc[0][j] = wmma_bf16(a0, b, acc[0][j]);
      acc[1][j] = wmma_bf16(a1, b, acc[1][j]);
    }
    stage_join();   // next buffer ready; everyone done with current
  }

  int grp = lane >> 4;
#pragma unroll
  for (int j = 0; j < 4; ++j) {
    int n = n0 + wn * 64 + j * 16 + (lane & 15);
    float bv = (flags & 1) ? bias[n] : 0.0f;
#pragma unroll
    for (int i = 0; i < 2; ++i)
#pragma unroll
      for (int r = 0; r < 8; ++r) {
        int m = m0 + wm * 32 + i * 16 + r + 8 * grp;
        float v = acc[i][j][r] + bv;
        if (flags & 2) v = 0.5f * v * (1.0f + erff(v * 0.70710678118654752f));
        v *= scale;
        if (flags & 8) {
          int bI = m >> 11, sI = m & 2047;     // token -> (batch, seq)
          int hI = n >> 6,  dI = n & 63;       // dim   -> (head, hd)
          ((__bf16*)C)[((((size_t)bI * NHEADS_ + hI) * HD_ + dI) << 11) + sI] =
              (__bf16)v;
        } else if (flags & 4) {
          ((__bf16*)C)[(size_t)m * N + n] = (__bf16)v;
        } else {
          ((float*)C)[(size_t)m * N + n] = v;
        }
      }
  }
}

// ---------------------------------------------------------------------------
// Fused attention, one (b, h, 32-query tile) per block (8 waves).
// Q,K bf16 [token][dim]; Vt bf16 [(b,h,d)][s]; ctx bf16 [token][dim].
// ---------------------------------------------------------------------------
__global__ __launch_bounds__(256) void attention_kernel(
    const __bf16* __restrict__ Q, const __bf16* __restrict__ Km,
    const __bf16* __restrict__ Vt, const int* __restrict__ mask,
    __bf16* __restrict__ ctx) {
  constexpr int QST = 72;     // sQ  [32][72]
  constexpr int KST = 72;     // sKV as K tile [key(128)][72]
  constexpr int VST = 136;    // sKV as V^T    [hd(64)][136]
  constexpr int PST = 2056;   // sP  [32][2056]

  __shared__ __bf16 sQ[32 * QST];
  __shared__ __bf16 sKV[128 * KST];     // 9216 elems; reused for V^T (8704)
  __shared__ __bf16 sP[32 * PST];
  __shared__ float  sInv[32];

  int tid  = threadIdx.x;
  int lane = tid & 31;
  int wid  = tid >> 5;
  int grp  = lane >> 4;

  int qt = blockIdx.x;
  int bh = blockIdx.y;
  int b  = bh >> 4;
  int h  = bh & 15;
  int qbase = qt * 32;
  size_t headoff = (size_t)b * SEQ_ * DIM_ + (size_t)h * HD_;

  // ---- stage Q tile (async; joined with first K tile) ----
  {
    int r = tid >> 3;           // 0..31
    int c = (tid & 7) * 8;      // 0..56
    cp16(Q + headoff + (size_t)(qbase + r) * DIM_ + c, &sQ[r * QST + c]);
  }

  // ---- phase 1: scores(32 x 2048) ----
  for (int kt = 0; kt < SEQ_ / 128; ++kt) {
    int kb = kt * 128;
    {
      int r  = tid >> 1;          // key-local 0..127
      int cc = (tid & 1) * 32;    // 0 or 32
      const __bf16* g = Km + headoff + (size_t)(kb + r) * DIM_ + cc;
#pragma unroll
      for (int j = 0; j < 4; ++j)
        cp16(g + j * 8, &sKV[r * KST + cc + j * 8]);
    }
    stage_join();

    int n0 = wid * 16;            // key chunk owned by this wave
    v8f s0 = {}, s1 = {};
    {
      v16bf b0  = frag_b(sKV, KST, n0, 0);
      v16bf b1  = frag_b(sKV, KST, n0, 32);
      v16bf a00 = frag_a(sQ, QST, 0,  0);
      v16bf a01 = frag_a(sQ, QST, 0,  32);
      v16bf a10 = frag_a(sQ, QST, 16, 0);
      v16bf a11 = frag_a(sQ, QST, 16, 32);
      s0 = wmma_bf16(a00, b0, s0);
      s0 = wmma_bf16(a01, b1, s0);
      s1 = wmma_bf16(a10, b0, s1);
      s1 = wmma_bf16(a11, b1, s1);
    }
    int key = kb + n0 + (lane & 15);
    int mk  = mask[b * SEQ_ + key];
#pragma unroll
    for (int r = 0; r < 8; ++r) {
      int m = r + 8 * grp;
      float v0 = mk ? s0[r] : -1e30f;
      float v1 = mk ? s1[r] : -1e30f;
      sP[(size_t)m * PST + key]        = (__bf16)v0;
      sP[(size_t)(m + 16) * PST + key] = (__bf16)v1;
    }
    __syncthreads();
  }

  // ---- phase 2: exact softmax, 4 rows per wave ----
#pragma unroll
  for (int rr = 0; rr < 4; ++rr) {
    int m = wid * 4 + rr;
    float mx = -3.0e38f;
    for (int c = lane; c < SEQ_; c += 32)
      mx = fmaxf(mx, (float)sP[(size_t)m * PST + c]);
#pragma unroll
    for (int o = 16; o > 0; o >>= 1) mx = fmaxf(mx, __shfl_xor(mx, o, 32));
    float sum = 0.0f;
    for (int c = lane; c < SEQ_; c += 32) {
      float p = __expf((float)sP[(size_t)m * PST + c] - mx);
      sP[(size_t)m * PST + c] = (__bf16)p;
      sum += p;
    }
#pragma unroll
    for (int o = 16; o > 0; o >>= 1) sum += __shfl_xor(sum, o, 32);
    if (lane == 0) sInv[m] = 1.0f / sum;
  }
  __syncthreads();

  // ---- phase 3: ctx = P @ V ----
  int mb = (wid >> 2) * 16;   // query rows
  int nb = (wid & 3) * 16;    // hd chunk
  v8f c0 = {};
  for (int kt = 0; kt < SEQ_ / 128; ++kt) {
    int kb = kt * 128;
    {
      int d  = tid >> 2;              // 0..63
      int cc = (tid & 3) * 32;        // key chunk 0,32,64,96
      const __bf16* g = Vt + ((size_t)bh * HD_ + d) * SEQ_ + kb + cc;
#pragma unroll
      for (int j = 0; j < 4; ++j)
        cp16(g + j * 8, &sKV[d * VST + cc + j * 8]);
    }
    stage_join();
#pragma unroll
    for (int kc = 0; kc < 128; kc += 32) {
      v16bf a  = frag_a(sP, PST, mb, kb + kc);
      v16bf bb = frag_b(sKV, VST, nb, kc);
      c0 = wmma_bf16(a, bb, c0);
    }
    __syncthreads();
  }
  {
    int n = nb + (lane & 15);
#pragma unroll
    for (int r = 0; r < 8; ++r) {
      int m = mb + r + 8 * grp;
      float v = c0[r] * sInv[m];
      ctx[headoff + (size_t)(qbase + m) * DIM_ + n] = (__bf16)v;
    }
  }
}

// ---------------------------------------------------------------------------
// outf = LayerNorm(A + B) * g + beta  (D=1024); optional bf16 duplicate.
// ---------------------------------------------------------------------------
__global__ __launch_bounds__(256) void ln_residual_kernel(
    const float* __restrict__ A, const float* __restrict__ B,
    const float* __restrict__ g, const float* __restrict__ be,
    float* __restrict__ outf, __bf16* __restrict__ outb) {
  __shared__ float rs[8], rq[8];
  int row = blockIdx.x;
  int tid = threadIdx.x;
  int lane = tid & 31;
  int wid  = tid >> 5;
  size_t base = (size_t)row * DIM_;

  float vals[4];
  float s = 0.0f, q = 0.0f;
#pragma unroll
  for (int k = 0; k < 4; ++k) {
    int i = k * 256 + tid;
    float t = A[base + i] + B[base + i];
    vals[k] = t;
    s += t;
    q += t * t;
  }
#pragma unroll
  for (int o = 16; o > 0; o >>= 1) {
    s += __shfl_xor(s, o, 32);
    q += __shfl_xor(q, o, 32);
  }
  if (lane == 0) { rs[wid] = s; rq[wid] = q; }
  __syncthreads();
  if (tid == 0) {
    float ts = 0.0f, tq = 0.0f;
#pragma unroll
    for (int i = 0; i < 8; ++i) { ts += rs[i]; tq += rq[i]; }
    rs[0] = ts; rq[0] = tq;
  }
  __syncthreads();
  float mean = rs[0] * (1.0f / DIM_);
  float var  = rq[0] * (1.0f / DIM_) - mean * mean;
  float inv  = rsqrtf(var + 1e-12f);
#pragma unroll
  for (int k = 0; k < 4; ++k) {
    int i = k * 256 + tid;
    float v = (vals[k] - mean) * inv * g[i] + be[i];
    outf[base + i] = v;
    if (outb) outb[base + i] = (__bf16)v;
  }
}

// ---------------------------------------------------------------------------
// Host-side orchestration
// ---------------------------------------------------------------------------
extern "C" void kernel_launch(void* const* d_in, const int* in_sizes, int n_in,
                              void* d_out, int out_size, void* d_ws, size_t ws_size,
                              hipStream_t stream) {
  (void)in_sizes; (void)n_in; (void)out_size; (void)ws_size;

  const float* x     = (const float*)d_in[0];
  const int*   amask = (const int*)d_in[1];
  const float* Wq    = (const float*)d_in[2];
  const float* bq    = (const float*)d_in[3];
  const float* Wk    = (const float*)d_in[4];
  const float* bk    = (const float*)d_in[5];
  const float* Wv    = (const float*)d_in[6];
  const float* bv    = (const float*)d_in[7];
  const float* Wo    = (const float*)d_in[8];
  const float* bo    = (const float*)d_in[9];
  const float* ln1_g = (const float*)d_in[10];
  const float* ln1_b = (const float*)d_in[11];
  const float* W1    = (const float*)d_in[12];
  const float* b1    = (const float*)d_in[13];
  const float* W2    = (const float*)d_in[14];
  const float* b2    = (const float*)d_in[15];
  const float* ln2_g = (const float*)d_in[16];
  const float* ln2_b = (const float*)d_in[17];

  const int M = 2 * SEQ_;   // 4096 token rows
  char* base = (char*)d_ws;
  const size_t MB = 1024ull * 1024ull;
  __bf16* xb   = (__bf16*)(base + 0 * MB);     //  8 MB
  __bf16* WqT  = (__bf16*)(base + 8 * MB);     //  2 MB each
  __bf16* WkT  = (__bf16*)(base + 10 * MB);
  __bf16* WvT  = (__bf16*)(base + 12 * MB);
  __bf16* WoT  = (__bf16*)(base + 14 * MB);
  __bf16* W1T  = (__bf16*)(base + 16 * MB);    //  8 MB
  __bf16* W2T  = (__bf16*)(base + 24 * MB);    //  8 MB
  float*  SAf  = (float*) (base + 32 * MB);    // 16 MB
  __bf16* SAb  = (__bf16*)(base + 48 * MB);    //  8 MB
  __bf16* Qb   = (__bf16*)(base + 56 * MB);    //  8 MB
  __bf16* Kb   = (__bf16*)(base + 64 * MB);    //  8 MB
  __bf16* VtB  = (__bf16*)(base + 72 * MB);    //  8 MB
  __bf16* CTXb = (__bf16*)(base + 80 * MB);    //  8 MB
  float*  OPf  = (float*) (base + 56 * MB);    // 16 MB (reuses Q+K, dead)
  __bf16* Hb   = (__bf16*)(base + 56 * MB);    // 32 MB (reuses Q..CTX, dead)
  float*  FFf  = (float*) (base + 88 * MB);    // 16 MB -> total 104 MB

  dim3 blk(256);

  // ---- one-time (per call) operand conversion ----
  cvt_bf16_kernel<<<dim3(4096), blk, 0, stream>>>(x, xb, M * DIM_);
  transpose_cvt_kernel<<<dim3(32, 32),  blk, 0, stream>>>(Wq, WqT, DIM_, DIM_);
  transpose_cvt_kernel<<<dim3(32, 32),  blk, 0, stream>>>(Wk, WkT, DIM_, DIM_);
  transpose_cvt_kernel<<<dim3(32, 32),  blk, 0, stream>>>(Wv, WvT, DIM_, DIM_);
  transpose_cvt_kernel<<<dim3(32, 32),  blk, 0, stream>>>(Wo, WoT, DIM_, DIM_);
  transpose_cvt_kernel<<<dim3(128, 32), blk, 0, stream>>>(W1, W1T, DIM_, 4096);
  transpose_cvt_kernel<<<dim3(32, 128), blk, 0, stream>>>(W2, W2T, 4096, DIM_);

  dim3 g_d(DIM_ / 256, M / 64);      // N=1024 GEMMs
  dim3 g_h(4096 / 256, M / 64);      // N=4096 GEMM
  dim3 g_attn(SEQ_ / 32, 2 * NHEADS_);

  // QKV projections (Q folds 1/sqrt(64); V written per-head transposed)
  gemm_bf16_kernel<<<g_d, blk, 0, stream>>>(xb, WqT, bq, Qb,  M, DIM_, DIM_, 1 | 4, 0.125f);
  gemm_bf16_kernel<<<g_d, blk, 0, stream>>>(xb, WkT, bk, Kb,  M, DIM_, DIM_, 1 | 4, 1.0f);
  gemm_bf16_kernel<<<g_d, blk, 0, stream>>>(xb, WvT, bv, VtB, M, DIM_, DIM_, 1 | 8, 1.0f);

  attention_kernel<<<g_attn, blk, 0, stream>>>(Qb, Kb, VtB, amask, CTXb);

  // output projection + residual LN (SA in f32 and bf16)
  gemm_bf16_kernel<<<g_d, blk, 0, stream>>>(CTXb, WoT, bo, OPf, M, DIM_, DIM_, 1, 1.0f);
  ln_residual_kernel<<<dim3(M), blk, 0, stream>>>(OPf, x, ln1_g, ln1_b, SAf, SAb);

  // FFN: gelu(SA @ W1 + b1) @ W2 + b2, residual LN -> d_out
  gemm_bf16_kernel<<<g_h, blk, 0, stream>>>(SAb, W1T, b1, Hb, M, 4096, DIM_, 1 | 2 | 4, 1.0f);
  gemm_bf16_kernel<<<g_d, blk, 0, stream>>>(Hb, W2T, b2, FFf, M, DIM_, 4096, 1, 1.0f);
  ln_residual_kernel<<<dim3(M), blk, 0, stream>>>(FFf, SAf, ln2_g, ln2_b, (float*)d_out, nullptr);
}
